// IntLinear_80599356277152
// MI455X (gfx1250) — 8-bit matrix operand order repaired and checked by reading, compiled, not run on hardware
//
#include <hip/hip_runtime.h>


typedef __attribute__((ext_vector_type(8))) int v8i;

#define MDIM 8192
#define NDIM 4096
#define KDIM 4096
#define KTILES (KDIM / 64)

__global__ void pack_a_int8(const float* __restrict__ src, int* __restrict__ dst,
                            long total_ints) {
  long o = (long)blockIdx.x * blockDim.x + threadIdx.x;
  if (o >= total_ints) return;
  int tile = (int)(o >> 8);
  int w    = (int)(o & 255);
  int lane = w >> 3;
  int v    = w & 7;
  int rt   = tile >> 6;
  int kt   = tile & (KTILES - 1);
  int row  = rt * 16 + (lane & 15);
  int half = lane >> 4;
  int k    = kt * 64 + ((v >> 1) << 4) + (half << 3) + ((v & 1) << 2);
  const float* p = src + (long)row * KDIM + k;
  int packed = 0;
#pragma unroll
  for (int b = 0; b < 4; ++b) {
    int vi = (int)p[b];
    vi = vi < -128 ? -128 : (vi > 127 ? 127 : vi);
    packed |= (vi & 0xFF) << (8 * b);
  }
  dst[o] = packed;
}

__global__ void pack_b_int8(const float* __restrict__ wgt, int* __restrict__ dst,
                            long total_ints) {
  long o = (long)blockIdx.x * blockDim.x + threadIdx.x;
  if (o >= total_ints) return;
  int tile = (int)(o >> 8);
  int w    = (int)(o & 255);
  int lane = w >> 3;
  int v    = w & 7;
  int nt   = tile >> 6;
  int kt   = tile & (KTILES - 1);
  int k    = kt * 64 + ((v >> 1) << 4) + ((lane >> 4) << 3) + ((v & 1) << 2);
  int n0   = nt * 16 + (lane & 15);
  int packed = 0;
#pragma unroll
  for (int b = 0; b < 4; ++b) {
    int vi = (int)wgt[(long)n0 * KDIM + k + b];
    vi = vi < -128 ? -128 : (vi > 127 ? 127 : vi);
    packed |= (vi & 0xFF) << (8 * b);
  }
  dst[o] = packed;
}

__global__ __launch_bounds__(256)
void gemm_iu8_kernel(const unsigned char* __restrict__ Ap,
                     const unsigned char* __restrict__ Bp,
                     const float* __restrict__ bias,
                     float* __restrict__ out) {
  const int lane = threadIdx.x & 31;
  const int wave = threadIdx.x >> 5;
  const int wm   = wave >> 2;
  const int wn   = wave & 3;
  const int mt0  = blockIdx.x * 8  + wm * 4;
  const int nt0  = blockIdx.y * 16 + wn * 4;

  v8i acc[4][4] = {};

  const v8i* __restrict__ Afrag = (const v8i*)Ap;
  const v8i* __restrict__ Bfrag = (const v8i*)Bp;

  for (int kt = 0; kt < KTILES; ++kt) {
    v8i a[4], b[4];
#pragma unroll
    for (int i = 0; i < 4; ++i)
      a[i] = Afrag[(long)((mt0 + i) * KTILES + kt) * 32 + lane];
#pragma unroll
    for (int j = 0; j < 4; ++j)
      b[j] = Bfrag[(long)((nt0 + j) * KTILES + kt) * 32 + lane];
#pragma unroll
    for (int i = 0; i < 4; ++i)
#pragma unroll
      for (int j = 0; j < 4; ++j)
        acc[i][j] = __builtin_amdgcn_wmma_i32_16x16x64_iu8(
             true, a[i],  true, b[j], acc[i][j],
             false,  false);
  }

  const int ncol = lane & 15;
  const int mrow = (lane >> 4) << 3;
#pragma unroll
  for (int j = 0; j < 4; ++j) {
    int n  = (nt0 + j) * 16 + ncol;
    int bv = (int)bias[n];
#pragma unroll
    for (int i = 0; i < 4; ++i) {
      long mbase = (long)((mt0 + i) * 16 + mrow);
#pragma unroll
      for (int r = 0; r < 8; ++r)
        out[(mbase + r) * NDIM + n] = (float)(acc[i][j][r] + bv);
    }
  }
}

extern "C" void kernel_launch(void* const* d_in, const int* in_sizes, int n_in,
                              void* d_out, int out_size, void* d_ws, size_t ws_size,
                              hipStream_t stream) {
  const float* x    = (const float*)d_in[0];
  const float* wgt  = (const float*)d_in[1];
  const float* bias = (const float*)d_in[2];
  float* out        = (float*)d_out;

  unsigned char* Ap = (unsigned char*)d_ws;
  unsigned char* Bp = Ap + (size_t)MDIM * KDIM;

  {
    long total = (long)MDIM * KDIM / 4;
    pack_a_int8<<<(int)((total + 255) / 256), 256, 0, stream>>>(x, (int*)Ap, total);
  }
  {
    long total = (long)NDIM * KDIM / 4;
    pack_b_int8<<<(int)((total + 255) / 256), 256, 0, stream>>>(wgt, (int*)Bp, total);
  }

  dim3 grid(MDIM / 128, NDIM / 256);
  gemm_iu8_kernel<<<grid, 256, 0, stream>>>(Ap, Bp, bias, out);
}
